// GNN_node_specific_395136991892
// MI455X (gfx1250) — compile-verified, matrix operand order checked by reading, not run on hardware
//
#include <hip/hip_runtime.h>
#include <hip/hip_bf16.h>

typedef __attribute__((ext_vector_type(16))) __bf16 v16bf;
typedef __attribute__((ext_vector_type(4)))  __bf16 v4bf;
typedef __attribute__((ext_vector_type(8)))  float  v8f;

#define N_NODES 16000
#define E_EDGES 512000
#define G_GENES 2000
#define R_ROWS  500000     // G * (N / 64)
#define C_OUT   128
#define BN_EPS  1e-5f

// ---------------------------------------------------------------- small prep
__global__ void k_zero(float* __restrict__ p, int n) {
  int i = blockIdx.x * blockDim.x + threadIdx.x;
  if (i < n) p[i] = 0.f;
}

__global__ void k_deg(const int* __restrict__ ei, const float* __restrict__ ea,
                      float* __restrict__ deg) {
  int e = blockIdx.x * blockDim.x + threadIdx.x;
  if (e < E_EDGES) atomicAdd(&deg[ei[E_EDGES + e]], ea[e]);
}

__global__ void k_dinv_agginit(const float* __restrict__ deg, const float* __restrict__ x,
                               float* __restrict__ dinv, float* __restrict__ agg) {
  int n = blockIdx.x * blockDim.x + threadIdx.x;
  if (n < N_NODES) {
    float di = rsqrtf(deg[n] + 1.0f);
    dinv[n] = di;
    float dd = di * di;
    agg[2*n]   = x[2*n]   * dd;
    agg[2*n+1] = x[2*n+1] * dd;
  }
}

__global__ void k_scatter(const int* __restrict__ ei, const float* __restrict__ ea,
                          const float* __restrict__ dinv, const float* __restrict__ x,
                          float* __restrict__ agg) {
  int e = blockIdx.x * blockDim.x + threadIdx.x;
  if (e < E_EDGES) {
    int s = ei[e], d = ei[E_EDGES + e];
    float w = dinv[s] * ea[e] * dinv[d];
    atomicAdd(&agg[2*d],   w * x[2*s]);
    atomicAdd(&agg[2*d+1], w * x[2*s+1]);
  }
}

// feats[g*16000 + n] = agg[n]·Wg + bg - x0[n]  == row-major [R_ROWS, 64] bf16
__global__ void k_feats(const float* __restrict__ agg, const float* __restrict__ x,
                        const float* __restrict__ gW, const float* __restrict__ gb,
                        __bf16* __restrict__ feats) {
  int n = blockIdx.x * blockDim.x + threadIdx.x;
  int g = blockIdx.y;
  if (n < N_NODES) {
    float w0 = gW[g * 128];            // gnn_W[g,0,0]
    float w1 = gW[g * 128 + 64];       // gnn_W[g,1,0]
    float bg = gb[g * 64];             // gnn_b[g,0]
    float v = agg[2*n] * w0 + agg[2*n+1] * w1 + bg - x[2*n];
    feats[(size_t)g * N_NODES + n] = (__bf16)v;
  }
}

// sum/sumsq -> affine BN coefficients: y = scale*h + shift
__global__ void k_bnfin(const float* __restrict__ sum, const float* __restrict__ sqs,
                        const float* __restrict__ g, const float* __restrict__ beta,
                        float* __restrict__ scale, float* __restrict__ shift) {
  int j = threadIdx.x;
  const float invR = 1.0f / (float)R_ROWS;
  float mu  = sum[j] * invR;
  float var = sqs[j] * invR - mu * mu;
  float is  = g[j] * rsqrtf(var + BN_EPS);
  scale[j] = is;
  shift[j] = beta[j] - mu * is;
}

// ------------------------------------------------------ fused GEMM + BN pass
// out[R,128] = (TRANS ? relu(iscale*in + ishift) : in)[R,KDIM] @ W[KDIM,128] + bias
// optional per-column sum / sum-of-squares for the *output* BN stats.
// Block: 160 threads = 5 waves; each wave: 16 rows x 128 cols; grid = 6250 (exact).
template <int KDIM, bool TRANS, bool STATS>
__global__ __launch_bounds__(160)
void k_gemm(const __bf16* __restrict__ in, const float* __restrict__ W,
            const float* __restrict__ bias,
            const float* __restrict__ iscale, const float* __restrict__ ishift,
            __bf16* __restrict__ out,
            float* __restrict__ osum, float* __restrict__ osqs) {
  __shared__ __bf16 sWt[C_OUT * KDIM];   // W transposed: sWt[n*KDIM + k]
  __shared__ float  sbias[C_OUT];
  __shared__ float  sscale[KDIM];
  __shared__ float  sshift[KDIM];
  __shared__ float  ssum[C_OUT];
  __shared__ float  ssq[C_OUT];

  const int tid = threadIdx.x;
  for (int i = tid; i < C_OUT * KDIM; i += 160) {
    int k = i % KDIM, n = i / KDIM;
    sWt[i] = (__bf16)W[k * C_OUT + n];
  }
  for (int i = tid; i < C_OUT; i += 160) {
    sbias[i] = bias[i];
    if (STATS) { ssum[i] = 0.f; ssq[i] = 0.f; }
  }
  if (TRANS) {
    for (int i = tid; i < KDIM; i += 160) { sscale[i] = iscale[i]; sshift[i] = ishift[i]; }
  }
  __syncthreads();

  const int lane = tid & 31;
  const int wave = tid >> 5;
  const int half = lane >> 4;          // 0: lanes 0-15, 1: lanes 16-31
  const int l16  = lane & 15;
  const size_t row_base = (size_t)blockIdx.x * 80 + (size_t)wave * 16;

  v8f acc[8];
  const v8f vzero = {0.f, 0.f, 0.f, 0.f, 0.f, 0.f, 0.f, 0.f};
#pragma unroll
  for (int t = 0; t < 8; ++t) acc[t] = vzero;

  // one base pointer for all A loads: per-kstep offsets are compile-time
  const __bf16* abase = in + (row_base + (size_t)l16) * KDIM + half * 16;

#pragma unroll
  for (int k0 = 0; k0 < KDIM; k0 += 32) {
    const int koff = k0 + half * 16;   // first K index this half-wave owns
    v16bf a;
    if (TRANS) {                       // apply input BN + ReLU on load
      v16bf raw = *(const v16bf*)(abase + k0);           // 2x global_load_b128
      const float4* s4 = (const float4*)&sscale[koff];   // 64B-aligned LDS
      const float4* h4 = (const float4*)&sshift[koff];
#pragma unroll
      for (int q = 0; q < 4; ++q) {
        float4 sc = s4[q], sh = h4[q];
#pragma unroll
        for (int r = 0; r < 4; ++r) {
          int i = q * 4 + r;
          float v = (float)raw[i];
          float scq = (r == 0) ? sc.x : (r == 1) ? sc.y : (r == 2) ? sc.z : sc.w;
          float shq = (r == 0) ? sh.x : (r == 1) ? sh.y : (r == 2) ? sh.z : sh.w;
          v = scq * v + shq;
          v = v > 0.f ? v : 0.f;
          a[i] = (__bf16)v;
        }
      }
    } else {
      a = *(const v16bf*)(abase + k0);                   // 2x global_load_b128
    }
#pragma unroll
    for (int t = 0; t < 8; ++t) {
      v16bf b = *(const v16bf*)&sWt[(t * 16 + l16) * KDIM + koff];
      acc[t] = __builtin_amdgcn_wmma_f32_16x16x32_bf16(
          false, a, false, b, (short)0, acc[t], false, false);
    }
  }

  // epilogue: bias, bf16 store (pre-BN), per-column stats.
  // Single base pointer; every store offset is a compile-time immediate.
  __bf16* op = out + (row_base + (size_t)(half * 8)) * C_OUT + l16;
#pragma unroll
  for (int t = 0; t < 8; ++t) {
    const int n = t * 16 + l16;
    const float bj = sbias[n];
    float cs = 0.f, cq = 0.f;
#pragma unroll
    for (int v = 0; v < 8; ++v) {
      float h = acc[t][v] + bj;
      op[v * C_OUT + t * 16] = (__bf16)h;   // global_store_b16, imm offset
      cs += h; cq += h * h;
    }
    if (STATS) { atomicAdd(&ssum[n], cs); atomicAdd(&ssq[n], cq); }
  }
  if (STATS) {
    __syncthreads();
    for (int i = tid; i < C_OUT; i += 160) {
      atomicAdd(&osum[i], ssum[i]);
      atomicAdd(&osqs[i], ssq[i]);
    }
  }
}

// ---------------------------------------------- final layer: BN+ReLU, dot 128->1
__global__ __launch_bounds__(256)
void k_final(const __bf16* __restrict__ in, const float* __restrict__ scale,
             const float* __restrict__ shift, const float* __restrict__ w,
             const float* __restrict__ b, float* __restrict__ out) {
  int lane = threadIdx.x & 31;
  int wave = threadIdx.x >> 5;
  size_t row = (size_t)blockIdx.x * 8 + wave;
  v4bf p = *(const v4bf*)(in + row * C_OUT + lane * 4);  // global_load_b64
  float s = 0.f;
#pragma unroll
  for (int i = 0; i < 4; ++i) {
    int j = lane * 4 + i;
    float v = (float)p[i];
    v = scale[j] * v + shift[j];
    v = v > 0.f ? v : 0.f;
    s += v * w[j];
  }
#pragma unroll
  for (int off = 16; off > 0; off >>= 1) s += __shfl_xor(s, off, 32);
  if (lane == 0) out[row] = s + b[0];
}

// ---------------------------------------------------------------- launcher
extern "C" void kernel_launch(void* const* d_in, const int* in_sizes, int n_in,
                              void* d_out, int out_size, void* d_ws, size_t ws_size,
                              hipStream_t stream) {
  const float* x    = (const float*)d_in[0];
  const int*   ei   = (const int*)d_in[1];
  const float* ea   = (const float*)d_in[2];
  const float* gW   = (const float*)d_in[3];
  const float* gb   = (const float*)d_in[4];
  const float* eW0  = (const float*)d_in[5];
  const float* eb0  = (const float*)d_in[6];
  const float* eg0  = (const float*)d_in[7];
  const float* ebt0 = (const float*)d_in[8];
  const float* eW1  = (const float*)d_in[9];
  const float* eb1  = (const float*)d_in[10];
  const float* eg1  = (const float*)d_in[11];
  const float* ebt1 = (const float*)d_in[12];
  const float* eW2  = (const float*)d_in[13];
  const float* eb2  = (const float*)d_in[14];
  const float* dW0  = (const float*)d_in[15];
  const float* db0  = (const float*)d_in[16];
  const float* dg0  = (const float*)d_in[17];
  const float* dbt0 = (const float*)d_in[18];
  const float* dW1  = (const float*)d_in[19];
  const float* db1  = (const float*)d_in[20];
  const float* dg1  = (const float*)d_in[21];
  const float* dbt1 = (const float*)d_in[22];
  const float* dW2  = (const float*)d_in[23];   // [128,1] -> flat [128]
  const float* db2  = (const float*)d_in[24];   // [1]

  char* ws = (char*)d_ws;
  __bf16* bufA = (__bf16*)ws;                         // 128 MB (feats uses first 64 MB)
  __bf16* bufB = (__bf16*)(ws + 128000000);           // 128 MB
  float*  smal = (float*)(ws + 256000000);            // graph + stats scratch
  float* deg  = smal;                  // 16000
  float* dinv = smal + 16000;          // 16000
  float* agg  = smal + 32000;          // 32000
  float* st   = smal + 64000;          // 4 layers x (sum[128], sqs[128]) = 1024
  float* sc   = smal + 64000 + 1024;   // 4 layers x (scale[128], shift[128]) = 1024
  const int nZero = 66048;

  // graph prep
  k_zero<<<(nZero + 255) / 256, 256, 0, stream>>>(smal, nZero);
  k_deg<<<(E_EDGES + 255) / 256, 256, 0, stream>>>(ei, ea, deg);
  k_dinv_agginit<<<(N_NODES + 255) / 256, 256, 0, stream>>>(deg, x, dinv, agg);
  k_scatter<<<(E_EDGES + 255) / 256, 256, 0, stream>>>(ei, ea, dinv, x, agg);
  k_feats<<<dim3((N_NODES + 255) / 256, G_GENES), 256, 0, stream>>>(agg, x, gW, gb, bufA);

  const int gemmGrid = R_ROWS / 80;   // 6250, exact

  // enc L0: feats(64) -> h0, stats s0
  k_gemm<64, false, true><<<gemmGrid, 160, 0, stream>>>(
      bufA, eW0, eb0, nullptr, nullptr, bufB, st + 0, st + 128);
  k_bnfin<<<1, 128, 0, stream>>>(st + 0, st + 128, eg0, ebt0, sc + 0, sc + 128);
  // enc L1: BN0+ReLU(h0) -> h1, stats s1
  k_gemm<128, true, true><<<gemmGrid, 160, 0, stream>>>(
      bufB, eW1, eb1, sc + 0, sc + 128, bufA, st + 256, st + 384);
  k_bnfin<<<1, 128, 0, stream>>>(st + 256, st + 384, eg1, ebt1, sc + 256, sc + 384);
  // enc L2: BN1+ReLU(h1) -> h2 (no BN after)
  k_gemm<128, true, false><<<gemmGrid, 160, 0, stream>>>(
      bufA, eW2, eb2, sc + 256, sc + 384, bufB, nullptr, nullptr);
  // dec L0: h2 -> h3, stats s2
  k_gemm<128, false, true><<<gemmGrid, 160, 0, stream>>>(
      bufB, dW0, db0, nullptr, nullptr, bufA, st + 512, st + 640);
  k_bnfin<<<1, 128, 0, stream>>>(st + 512, st + 640, dg0, dbt0, sc + 512, sc + 640);
  // dec L1: BN2+ReLU(h3) -> h4, stats s3
  k_gemm<128, true, true><<<gemmGrid, 160, 0, stream>>>(
      bufA, dW1, db1, sc + 512, sc + 640, bufB, st + 768, st + 896);
  k_bnfin<<<1, 128, 0, stream>>>(st + 768, st + 896, dg1, dbt1, sc + 768, sc + 896);
  // dec L2: BN3+ReLU(h4) . w -> out [500000]
  k_final<<<R_ROWS / 8, 256, 0, stream>>>(bufB, sc + 768, sc + 896, dW2, db2, (float*)d_out);

  (void)in_sizes; (void)n_in; (void)out_size; (void)ws_size;
}
